// TritterAttention_41197326303410
// MI455X (gfx1250) — matrix operand order repaired and checked by reading, not compiled, not run on hardware
//
#include <hip/hip_runtime.h>
#include <hip/hip_bf16.h>
#include <math.h>

// ---------------------------------------------------------------------------
// Types (clang ext vectors only; safe in unions)
// ---------------------------------------------------------------------------
typedef unsigned short u16;
typedef unsigned int   u32;
typedef __attribute__((ext_vector_type(2)))  u32    u32x2;
typedef __attribute__((ext_vector_type(4)))  u32    u32x4;
typedef __attribute__((ext_vector_type(4)))  float  f32x4;
typedef __attribute__((ext_vector_type(16))) __bf16 v16bf;
typedef __attribute__((ext_vector_type(8)))  float  v8f;

#define B_    2
#define S_    2048
#define H_    2048
#define NH_   16
#define HD_   128
#define MROWS (B_ * S_)   // 4096 token rows

// ---------------------------------------------------------------------------
// gfx1250 async global->LDS path (ASYNCcnt), gated so compile never regresses.
// Builtin prototype (from hipcc diagnostic): first param is a pointer to a
// gcc-style vector_size(16) int4 in the __device__ (AS1) address space.
// ---------------------------------------------------------------------------
#if defined(__HIP_DEVICE_COMPILE__) && \
    __has_builtin(__builtin_amdgcn_global_load_async_to_lds_b128)
#define ASYNC_LDS 1
#else
#define ASYNC_LDS 0
#endif

#if ASYNC_LDS
typedef int i32x4_g __attribute__((vector_size(16)));
typedef __attribute__((address_space(1))) i32x4_g as1_b128;  // global int4
typedef __attribute__((address_space(3))) i32x4_g as3_b128;  // LDS int4
__device__ __forceinline__ void async_cp16(const void* g, void* l) {
  // 16B per lane, global -> LDS, tracked by ASYNCcnt
  __builtin_amdgcn_global_load_async_to_lds_b128(
      (as1_b128*)g, (as3_b128*)l, 0, 0);
}
__device__ __forceinline__ void async_wait0() {
#if __has_builtin(__builtin_amdgcn_s_wait_asynccnt)
  __builtin_amdgcn_s_wait_asynccnt(0);
#else
  asm volatile("s_wait_asynccnt 0x0" ::: "memory");
#endif
}
#endif

// fp32 -> bf16, round-to-nearest-even
__device__ __forceinline__ u16 f2bf(float f) {
  u32 u = __float_as_uint(f);
  u += 0x7FFFu + ((u >> 16) & 1u);
  return (u16)(u >> 16);
}

// Build a v16bf WMMA fragment from two 16-byte LDS chunks
__device__ __forceinline__ v16bf ld_frag(const u16* lo, const u16* hi) {
  union { u32x4 u[2]; v16bf v; } t;
  t.u[0] = *(const u32x4*)lo;
  t.u[1] = *(const u32x4*)hi;
  return t.v;
}

__device__ __forceinline__ v8f wmma_bf16(v16bf a, v16bf b, v8f c) {
  // D = A(16x32) * B(32x16) + C, fp32 accumulate
  return __builtin_amdgcn_wmma_f32_16x16x32_bf16(
      /*neg_a=*/false, a, /*neg_b=*/false, b,
      /*c_mod=*/(short)0, c, /*reuse_a=*/false, /*reuse_b=*/false);
}

// ---------------------------------------------------------------------------
// Kernel 1: elementwise fp32 -> bf16 (vectorized by 4)
// ---------------------------------------------------------------------------
__global__ __launch_bounds__(256) void cvt_f32_bf16(
    const float* __restrict__ in, u16* __restrict__ out, int n4) {
  int i = blockIdx.x * 256 + threadIdx.x;
  if (i >= n4) return;
  f32x4 v = *((const f32x4*)in + i);
  u32x2 o;
  o.x = (u32)f2bf(v.x) | ((u32)f2bf(v.y) << 16);
  o.y = (u32)f2bf(v.z) | ((u32)f2bf(v.w) << 16);
  *((u32x2*)out + i) = o;
}

// ---------------------------------------------------------------------------
// Kernel 2: C[M,N] = A[M,K](bf16) * W[N,K](bf16)^T + bias[N]   (fp32 out)
// 128x128 block tile, 8 waves, each wave a 64x32 tile (4x2 WMMA accs).
// Tiles staged via async global->LDS when available; next tile prefetched.
// ---------------------------------------------------------------------------
#define GTM 128
#define GTK 32
#define GLDA 40   // padded LDS row stride in bf16 elems (80B, 16B aligned)

__global__ __launch_bounds__(256) void gemm_bf16_btrans(
    const u16* __restrict__ A, const u16* __restrict__ W,
    const float* __restrict__ bias, float* __restrict__ C,
    int M, int N, int Kdim) {
  __shared__ __align__(16) u16 As[GTM * GLDA];
  __shared__ __align__(16) u16 Ws[GTM * GLDA];

  const int tid  = threadIdx.x;
  const int lane = tid & 31;
  const int wid  = tid >> 5;
  const int wm   = wid >> 2;      // 0..1  (64 rows each)
  const int wn   = wid & 3;       // 0..3  (32 cols each)
  const int rl   = lane & 15;
  const int hs   = lane >> 4;
  const int m0   = blockIdx.y * 128;
  const int n0   = blockIdx.x * 128;
  const int r    = tid >> 1;      // 0..127 : LDS-stage row
  const int half = tid & 1;       // which 16-elem chunk of the 32-wide tile

  const u16* gA = &A[(size_t)(m0 + r) * Kdim + half * 16];
  const u16* gW = &W[(size_t)(n0 + r) * Kdim + half * 16];
  u16* lA = &As[r * GLDA + half * 16];
  u16* lW = &Ws[r * GLDA + half * 16];

  v8f zero = {};
  v8f acc[4][2];
#pragma unroll
  for (int i = 0; i < 4; ++i)
#pragma unroll
    for (int j = 0; j < 2; ++j) acc[i][j] = zero;

  for (int k0 = 0; k0 < Kdim; k0 += GTK) {
#if ASYNC_LDS
    async_cp16(gA + k0, lA);
    async_cp16(gW + k0, lW);
#else
    *(u32x4*)lA = *(const u32x4*)(gA + k0);
    *(u32x4*)lW = *(const u32x4*)(gW + k0);
#endif
    if (k0 + GTK < Kdim) {  // pull next K-tile toward the caches
      __builtin_prefetch(gA + k0 + GTK, 0, 1);
      __builtin_prefetch(gW + k0 + GTK, 0, 1);
    }
#if ASYNC_LDS
    async_wait0();
#endif
    __syncthreads();

    v16bf af[4], wf[2];
#pragma unroll
    for (int ti = 0; ti < 4; ++ti) {
      // A layout: lanes 0-15 hold K=[0..7,16..23], lanes 16-31 K=[8..15,24..31]
      const u16* p = &As[(wm * 64 + ti * 16 + rl) * GLDA + hs * 8];
      af[ti] = ld_frag(p, p + 16);
    }
#pragma unroll
    for (int tj = 0; tj < 2; ++tj) {
      // B layout: lanes 0-15 hold K=[0..7,16..23], lanes 16-31 K=[8..15,24..31]
      const u16* p = &Ws[(wn * 32 + tj * 16 + rl) * GLDA + hs * 8];
      wf[tj] = ld_frag(p, p + 16);
    }
#pragma unroll
    for (int ti = 0; ti < 4; ++ti)
#pragma unroll
      for (int tj = 0; tj < 2; ++tj)
        acc[ti][tj] = wmma_bf16(af[ti], wf[tj], acc[ti][tj]);
    __syncthreads();
  }

  // Epilogue: C/D layout -> lanes<16 rows 0-7, lanes>=16 rows 8-15, col=lane&15
#pragma unroll
  for (int ti = 0; ti < 4; ++ti) {
#pragma unroll
    for (int tj = 0; tj < 2; ++tj) {
      const int col  = n0 + wn * 32 + tj * 16 + rl;
      const float bv = bias[col];
      const int rowb = m0 + wm * 64 + ti * 16 + hs * 8;
#pragma unroll
      for (int rr = 0; rr < 8; ++rr)
        C[(size_t)(rowb + rr) * N + col] = acc[ti][tj][rr] + bv;
    }
  }
}

// ---------------------------------------------------------------------------
// Kernel 3: per-head-row LayerNorm (row length HD_=128), fp32 in, bf16 out.
// One wave per row: 32 lanes x 4 floats, shfl_xor reductions (wave32).
// ---------------------------------------------------------------------------
__global__ __launch_bounds__(256) void ln_head_rows(
    const float* __restrict__ in, u16* __restrict__ out,
    const float* __restrict__ gamma, const float* __restrict__ beta,
    int nrows) {
  const int row = blockIdx.x * 8 + (threadIdx.x >> 5);
  if (row >= nrows) return;
  const int lane = threadIdx.x & 31;

  f32x4 x = *((const f32x4*)(in + (size_t)row * HD_) + lane);
  float s = x.x + x.y + x.z + x.w;
#pragma unroll
  for (int off = 16; off; off >>= 1) s += __shfl_xor(s, off, 32);
  const float mu = s * (1.0f / HD_);

  f32x4 d;
  d.x = x.x - mu; d.y = x.y - mu; d.z = x.z - mu; d.w = x.w - mu;
  float vs = d.x * d.x + d.y * d.y + d.z * d.z + d.w * d.w;
#pragma unroll
  for (int off = 16; off; off >>= 1) vs += __shfl_xor(vs, off, 32);
  const float rstd = rsqrtf(vs * (1.0f / HD_) + 1e-5f);

  f32x4 g  = *((const f32x4*)gamma + lane);
  f32x4 bt = *((const f32x4*)beta + lane);
  float y0 = d.x * rstd * g.x + bt.x;
  float y1 = d.y * rstd * g.y + bt.y;
  float y2 = d.z * rstd * g.z + bt.z;
  float y3 = d.w * rstd * g.w + bt.w;

  u32x2 o;
  o.x = (u32)f2bf(y0) | ((u32)f2bf(y1) << 16);
  o.y = (u32)f2bf(y2) | ((u32)f2bf(y3) << 16);
  *((u32x2*)(out + (size_t)row * HD_) + lane) = o;
}

// ---------------------------------------------------------------------------
// Kernel 4: causal FlashAttention, bf16 WMMA, fp32 softmax state.
// Grid (S/64, NH, B); 128 threads = 4 waves; each wave owns 16 q rows.
// K tile staged async->LDS; V tile transposed through VGPRs; next KV tile
// prefetched while the current one is consumed by WMMAs.
// ---------------------------------------------------------------------------
__global__ __launch_bounds__(128) void flash_attn_bf16(
    const u16* __restrict__ Q, const u16* __restrict__ K,
    const u16* __restrict__ V, u16* __restrict__ O) {
  __shared__ __align__(16) u16 Qs[64 * 136];
  __shared__ __align__(16) u16 Ks[64 * 136];
  __shared__ __align__(16) u16 Vt[128 * 72];
  __shared__ __align__(16) u16 Ps[4][16 * 72];

  const int tid  = threadIdx.x;
  const int lane = tid & 31;
  const int w    = tid >> 5;     // wave id 0..3
  const int rl   = lane & 15;
  const int hs   = lane >> 4;
  const int q0   = blockIdx.x * 64;
  const int h    = blockIdx.y;
  const int b    = blockIdx.z;
  const size_t rstride = (size_t)NH_ * HD_;            // 2048 elems per token
  const size_t base    = (size_t)b * S_ * rstride + (size_t)h * HD_;

  // ---- stage Q tile (64 rows x 128) ----
  for (int i = tid; i < 64 * 16; i += 128) {
    const int rr = i >> 4, c8 = (i & 15) * 8;
#if ASYNC_LDS
    async_cp16(&Q[base + (size_t)(q0 + rr) * rstride + c8], &Qs[rr * 136 + c8]);
#else
    *(u32x4*)&Qs[rr * 136 + c8] =
        *(const u32x4*)&Q[base + (size_t)(q0 + rr) * rstride + c8];
#endif
  }
#if ASYNC_LDS
  async_wait0();
#endif
  __syncthreads();

  // Q fragments live in registers for the whole kernel (4 K-chunks of 32)
  v16bf qf[4];
#pragma unroll
  for (int kc = 0; kc < 4; ++kc) {
    const u16* p = &Qs[(w * 16 + rl) * 136 + kc * 32 + hs * 8];
    qf[kc] = ld_frag(p, p + 16);
  }

  v8f zero = {};
  float mrow[8], lrow[8];
  v8f o[8];
#pragma unroll
  for (int i = 0; i < 8; ++i) { mrow[i] = -1e30f; lrow[i] = 0.0f; o[i] = zero; }

  const float scale = 0.08838834764831845f;  // 1/sqrt(128)
  const int nkt = blockIdx.x + 1;            // causal: only tiles with k0 <= q0

  for (int kt = 0; kt < nkt; ++kt) {
    __syncthreads();  // previous tile's LDS readers done before restage

    // ---- stage K tile (async) + transposed V tile ----
    for (int i = tid; i < 64 * 16; i += 128) {
      const int rr = i >> 4, c8 = (i & 15) * 8;
      const size_t grow = base + (size_t)(kt * 64 + rr) * rstride + c8;
#if ASYNC_LDS
      async_cp16(&K[grow], &Ks[rr * 136 + c8]);
#else
      *(u32x4*)&Ks[rr * 136 + c8] = *(const u32x4*)&K[grow];
#endif
      u32x4 dv = *(const u32x4*)&V[grow];
      const u16* de = (const u16*)&dv;
#pragma unroll
      for (int j = 0; j < 8; ++j) Vt[(c8 + j) * 72 + rr] = de[j];
      if (kt + 1 < nkt) {  // prefetch next KV tile
        __builtin_prefetch(&K[grow + 64 * rstride], 0, 1);
        __builtin_prefetch(&V[grow + 64 * rstride], 0, 1);
      }
    }
#if ASYNC_LDS
    async_wait0();
#endif
    __syncthreads();

    // ---- scores S = Q K^T (16 q-rows x 64 keys per wave) ----
    v8f sc[4];
#pragma unroll
    for (int tj = 0; tj < 4; ++tj) {
      v8f a = zero;
#pragma unroll
      for (int kc = 0; kc < 4; ++kc) {
        const u16* p = &Ks[(tj * 16 + rl) * 136 + kc * 32 + hs * 8];
        a = wmma_bf16(qf[kc], ld_frag(p, p + 16), a);
      }
      sc[tj] = a;
    }

    // ---- scale + causal mask + online softmax (in D-layout) ----
    float mx[8];
#pragma unroll
    for (int r2 = 0; r2 < 8; ++r2) mx[r2] = -1e30f;
#pragma unroll
    for (int tj = 0; tj < 4; ++tj) {
      const int key = kt * 64 + tj * 16 + rl;
#pragma unroll
      for (int r2 = 0; r2 < 8; ++r2) {
        const int qr = q0 + w * 16 + hs * 8 + r2;
        float v = sc[tj][r2] * scale;
        v = (key <= qr) ? v : -1e30f;
        sc[tj][r2] = v;
        mx[r2] = fmaxf(mx[r2], v);
      }
    }
#pragma unroll
    for (int r2 = 0; r2 < 8; ++r2) {
#pragma unroll
      for (int off = 1; off < 16; off <<= 1)
        mx[r2] = fmaxf(mx[r2], __shfl_xor(mx[r2], off, 16));
      const float mn   = fmaxf(mrow[r2], mx[r2]);
      const float corr = __expf(mrow[r2] - mn);
      mrow[r2] = mn;
      lrow[r2] *= corr;
#pragma unroll
      for (int nt = 0; nt < 8; ++nt) o[nt][r2] *= corr;
    }
    float rs[8];
#pragma unroll
    for (int r2 = 0; r2 < 8; ++r2) rs[r2] = 0.0f;
#pragma unroll
    for (int tj = 0; tj < 4; ++tj)
#pragma unroll
      for (int r2 = 0; r2 < 8; ++r2) {
        const float p = __expf(sc[tj][r2] - mrow[r2]);
        sc[tj][r2] = p;
        rs[r2] += p;
      }
#pragma unroll
    for (int r2 = 0; r2 < 8; ++r2) {
#pragma unroll
      for (int off = 1; off < 16; off <<= 1)
        rs[r2] += __shfl_xor(rs[r2], off, 16);
      lrow[r2] += rs[r2];
    }

    // ---- P (D-layout) -> per-wave LDS -> A-fragment layout ----
    u16* pp = &Ps[w][0];
#pragma unroll
    for (int tj = 0; tj < 4; ++tj)
#pragma unroll
      for (int r2 = 0; r2 < 8; ++r2)
        pp[(hs * 8 + r2) * 72 + tj * 16 + rl] = f2bf(sc[tj][r2]);
    // same-wave DS ops complete in order; no block barrier needed here

    v16bf pf[2];
#pragma unroll
    for (int kc2 = 0; kc2 < 2; ++kc2) {
      const u16* p = &pp[rl * 72 + kc2 * 32 + hs * 8];
      pf[kc2] = ld_frag(p, p + 16);
    }
    // ---- O += P * V  (B fragments read rows of V^T: contiguous keys) ----
#pragma unroll
    for (int nt = 0; nt < 8; ++nt)
#pragma unroll
      for (int kc2 = 0; kc2 < 2; ++kc2) {
        const u16* p = &Vt[(nt * 16 + rl) * 72 + kc2 * 32 + hs * 8];
        o[nt] = wmma_bf16(pf[kc2], ld_frag(p, p + 16), o[nt]);
      }
  }

  // ---- normalize and store bf16 output ----
#pragma unroll
  for (int r2 = 0; r2 < 8; ++r2) {
    const float inv = 1.0f / lrow[r2];
#pragma unroll
    for (int nt = 0; nt < 8; ++nt) o[nt][r2] *= inv;
  }
#pragma unroll
  for (int nt = 0; nt < 8; ++nt) {
    const int col = h * HD_ + nt * 16 + rl;
#pragma unroll
    for (int r2 = 0; r2 < 8; ++r2) {
      const size_t orow = (size_t)b * S_ + q0 + w * 16 + hs * 8 + r2;
      O[orow * rstride + col] = f2bf(o[nt][r2]);
    }
  }
}

// ---------------------------------------------------------------------------
// Host-side launch
// ---------------------------------------------------------------------------
extern "C" void kernel_launch(void* const* d_in, const int* in_sizes, int n_in,
                              void* d_out, int out_size, void* d_ws,
                              size_t ws_size, hipStream_t stream) {
  (void)in_sizes; (void)n_in; (void)out_size; (void)ws_size;

  const float* hidden = (const float*)d_in[0];
  const float* Wq  = (const float*)d_in[1];
  const float* bq  = (const float*)d_in[2];
  const float* Wk  = (const float*)d_in[3];
  const float* bk  = (const float*)d_in[4];
  const float* Wv  = (const float*)d_in[5];
  const float* bv  = (const float*)d_in[6];
  const float* Wo  = (const float*)d_in[7];
  const float* bo  = (const float*)d_in[8];
  const float* qg  = (const float*)d_in[9];
  const float* qbt = (const float*)d_in[10];
  const float* kg  = (const float*)d_in[11];
  const float* kbt = (const float*)d_in[12];
  float* out = (float*)d_out;

  // Workspace carve-up (~208 MiB total)
  char* ws = (char*)d_ws;
  const size_t szXb = (size_t)MROWS * H_ * sizeof(u16);  // 16 MiB
  const size_t szWb = (size_t)H_ * H_ * sizeof(u16);     //  8 MiB
  const size_t szQf = (size_t)MROWS * H_ * sizeof(float);// 32 MiB
  u16*  hXb = (u16*)ws;  ws += szXb;
  u16*  Wqb = (u16*)ws;  ws += szWb;
  u16*  Wkb = (u16*)ws;  ws += szWb;
  u16*  Wvb = (u16*)ws;  ws += szWb;
  u16*  Wob = (u16*)ws;  ws += szWb;
  float* Qf = (float*)ws; ws += szQf;
  float* Kf = (float*)ws; ws += szQf;
  float* Vf = (float*)ws; ws += szQf;
  u16*  Qb  = (u16*)ws;  ws += szXb;
  u16*  Kb  = (u16*)ws;  ws += szXb;
  u16*  Vb  = (u16*)ws;  ws += szXb;
  u16*  AOb = (u16*)ws;  ws += szXb;

  // 1) fp32 -> bf16 converts (activations + weights)
  const int n4x = MROWS * H_ / 4;
  const int n4w = H_ * H_ / 4;
  cvt_f32_bf16<<<(n4x + 255) / 256, 256, 0, stream>>>(hidden, hXb, n4x);
  cvt_f32_bf16<<<(n4w + 255) / 256, 256, 0, stream>>>(Wq, Wqb, n4w);
  cvt_f32_bf16<<<(n4w + 255) / 256, 256, 0, stream>>>(Wk, Wkb, n4w);
  cvt_f32_bf16<<<(n4w + 255) / 256, 256, 0, stream>>>(Wv, Wvb, n4w);
  cvt_f32_bf16<<<(n4w + 255) / 256, 256, 0, stream>>>(Wo, Wob, n4w);

  // 2) QKV projections: X @ W^T + b  (fp32 accumulate/out)
  dim3 gg(H_ / 128, MROWS / 128);
  gemm_bf16_btrans<<<gg, 256, 0, stream>>>(hXb, Wqb, bq, Qf, MROWS, H_, H_);
  gemm_bf16_btrans<<<gg, 256, 0, stream>>>(hXb, Wkb, bk, Kf, MROWS, H_, H_);
  gemm_bf16_btrans<<<gg, 256, 0, stream>>>(hXb, Wvb, bv, Vf, MROWS, H_, H_);

  // 3) per-head QK LayerNorm -> bf16; V plain convert -> bf16
  const int lnrows = B_ * S_ * NH_;  // 65536 rows of length 128
  ln_head_rows<<<lnrows / 8, 256, 0, stream>>>(Qf, Qb, qg, qbt, lnrows);
  ln_head_rows<<<lnrows / 8, 256, 0, stream>>>(Kf, Kb, kg, kbt, lnrows);
  cvt_f32_bf16<<<(n4x + 255) / 256, 256, 0, stream>>>(Vf, Vb, n4x);

  // 4) causal flash attention
  dim3 ga(S_ / 64, NH_, B_);
  flash_attn_bf16<<<ga, 128, 0, stream>>>(Qb, Kb, Vb, AOb);

  // 5) output projection -> fp32 d_out
  gemm_bf16_btrans<<<gg, 256, 0, stream>>>(AOb, Wob, bo, out, MROWS, H_, H_);
}
